// ParallelMLayer_47064251630010
// MI455X (gfx1250) — matrix operand order repaired and checked by reading, not compiled, not run on hardware
//
#include <hip/hip_runtime.h>
#include <hip/hip_bf16.h>
#include <math.h>

// ---------------- problem constants (match reference) ----------------
#define DIM    1024
#define HEADS  8
#define BATCH  4
#define SEQ    4096
#define MROWS  (BATCH * SEQ)       // 16384 rows of the GEMMs
#define BASE_DT 0.1f

#define NSEG   16
#define LSEG   (SEQ / NSEG)        // 256
#define NCH    (BATCH * DIM)       // 4096 independent scan channels

#define LDSROW 40                  // 32 K-elems + 8 pad (breaks bank conflicts)

typedef __attribute__((ext_vector_type(16))) __bf16 v16bf;
typedef __attribute__((ext_vector_type(8)))  __bf16 v8bf;
typedef __attribute__((ext_vector_type(4)))  __bf16 v4bf;
typedef __attribute__((ext_vector_type(8)))  float  v8f;

// low 32 bits of a generic pointer to __shared__ = LDS byte offset
__device__ __forceinline__ unsigned lds_off(const void* p) {
  return (unsigned)(size_t)p;
}

// gfx1250 async copy global -> LDS, 16 bytes per lane, tracked by ASYNCcnt
#define ASYNC_B128(ldsu32, gptr)                                        \
  asm volatile("global_load_async_to_lds_b128 %0, %1, off"              \
               :: "v"(ldsu32), "v"((const void*)(gptr)) : "memory")

// ---------------- fp32 -> bf16 conversion (vectorized, grid-stride) ----
__global__ void pm_cvt_bf16(const float4* __restrict__ in,
                            v4bf* __restrict__ out, int n4) {
  int i = blockIdx.x * blockDim.x + threadIdx.x;
  int stride = gridDim.x * blockDim.x;
  for (; i < n4; i += stride) {
    float4 f = in[i];
    v4bf o = { (__bf16)f.x, (__bf16)f.y, (__bf16)f.z, (__bf16)f.w };
    out[i] = o;
  }
}

// ---------------- fused triple GEMM (WMMA bf16 -> f32) + epilogue ------
// Block tile: 64(M) x 64(N), 8 waves as 2(M) x 4(N), each wave 32x16 per
// weight (2 M-subtiles), 6 accumulators. A/B tiles double-buffered in LDS
// via global_load_async_to_lds_b128 (ASYNCcnt ping-pong).
__global__ __launch_bounds__(256)
void pm_gemm3_epilogue(const __bf16* __restrict__ F,    // [MROWS, DIM]
                       const __bf16* __restrict__ Wa,   // [DIM, DIM]
                       const __bf16* __restrict__ Wb,
                       const __bf16* __restrict__ Wd,
                       const float* __restrict__ bA,
                       const float* __restrict__ bB,
                       const float* __restrict__ bdt,
                       float* __restrict__ Ag,          // [MROWS, DIM]
                       float* __restrict__ Bg,
                       float* __restrict__ Dg) {
  __shared__ __bf16 Abuf[2][64][LDSROW];       // 10 KB
  __shared__ __bf16 Bbuf[2][3][64][LDSROW];    // 30 KB

  const int tid   = threadIdx.x;
  const int lane  = tid & 31;
  const int wave  = tid >> 5;
  const int waveM = wave & 1;          // 2 waves along M
  const int waveN = wave >> 1;         // 4 waves along N
  const int m_blk = blockIdx.y * 64;
  const int n_blk = blockIdx.x * 64;

  // ---- staging: each thread owns one 16B chunk of each tile ----
  const int lrow   = tid >> 2;             // 0..63
  const int lchunk = (tid & 3) * 8;        // 0,8,16,24  (bf16 elems)
  const __bf16* gA  = F  + (size_t)(m_blk + lrow) * DIM + lchunk;
  const __bf16* gWa = Wa + (size_t)(n_blk + lrow) * DIM + lchunk;
  const __bf16* gWb = Wb + (size_t)(n_blk + lrow) * DIM + lchunk;
  const __bf16* gWd = Wd + (size_t)(n_blk + lrow) * DIM + lchunk;
  unsigned lA[2], lBa[2], lBb[2], lBd[2];
#pragma unroll
  for (int q = 0; q < 2; ++q) {
    lA[q]  = lds_off(&Abuf[q][lrow][lchunk]);
    lBa[q] = lds_off(&Bbuf[q][0][lrow][lchunk]);
    lBb[q] = lds_off(&Bbuf[q][1][lrow][lchunk]);
    lBd[q] = lds_off(&Bbuf[q][2][lrow][lchunk]);
  }
  auto issue = [&](int q, int k0) {
    ASYNC_B128(lA[q],  gA  + k0);
    ASYNC_B128(lBa[q], gWa + k0);
    ASYNC_B128(lBb[q], gWb + k0);
    ASYNC_B128(lBd[q], gWd + k0);
  };

  // ---- compute-side fragment coordinates (ISA fragment layouts) ----
  const int rc   = lane & 15;          // A row / B col within 16-tile
  const int half = lane >> 4;          // K-half selector
  const int ar0  = waveM * 32 + rc;    // A rows for the two M-subtiles
  const int ar1  = ar0 + 16;
  const int brow = waveN * 16 + rc;    // B row (= output column)

  v8f acc[3][2];
#pragma unroll
  for (int w = 0; w < 3; ++w)
#pragma unroll
    for (int mt = 0; mt < 2; ++mt) acc[w][mt] = (v8f){};

  issue(0, 0);                                   // prologue: stage K-step 0
  int buf = 0;
  for (int k0 = 0; k0 < DIM; k0 += 32) {
    if (k0 + 32 < DIM) {
      issue(buf ^ 1, k0 + 32);                   // stage next while computing
      asm volatile("s_wait_asynccnt 0x4" ::: "memory");  // prev 4-op set done
    } else {
      asm volatile("s_wait_asynccnt 0x0" ::: "memory");
    }
    __syncthreads();                             // LDS writes visible to all

    // A fragments (16x32 bf16): lane holds 16 contiguous K {half*16..+15}
    v8bf lo0 = *(const v8bf*)&Abuf[buf][ar0][half * 16];
    v8bf hi0 = *(const v8bf*)&Abuf[buf][ar0][half * 16 + 8];
    v8bf lo1 = *(const v8bf*)&Abuf[buf][ar1][half * 16];
    v8bf hi1 = *(const v8bf*)&Abuf[buf][ar1][half * 16 + 8];
    v16bf a0, a1;
#pragma unroll
    for (int i = 0; i < 8; ++i) {
      a0[i] = lo0[i]; a0[8 + i] = hi0[i];
      a1[i] = lo1[i]; a1[8 + i] = hi1[i];
    }
    // B fragments (32x16 bf16): lane holds 16 contiguous K of output col n
#pragma unroll
    for (int w = 0; w < 3; ++w) {
      v16bf bw = *(const v16bf*)&Bbuf[buf][w][brow][half * 16];
      acc[w][0] = __builtin_amdgcn_wmma_f32_16x16x32_bf16(
          false, a0, false, bw, (short)0, acc[w][0], false, false);
      acc[w][1] = __builtin_amdgcn_wmma_f32_16x16x32_bf16(
          false, a1, false, bw, (short)0, acc[w][1], false, false);
    }
    __syncthreads();                             // done reading this buffer
    buf ^= 1;
  }

  // ---- epilogue: gates. C/D layout: VGPR r -> m = r + 8*half, n = lane&15
  const int n = n_blk + waveN * 16 + rc;
  const float biasA = bA[n], biasB = bB[n], biasD = bdt[n];
  const float scale = exp2f((float)(n >> 7) * 0.58496250072115618f); // 1.5^head

#pragma unroll
  for (int mt = 0; mt < 2; ++mt) {
#pragma unroll
    for (int r = 0; r < 8; ++r) {
      const int m = m_blk + waveM * 32 + mt * 16 + r + half * 8;
      const size_t idx = (size_t)m * DIM + n;
      const float ga = acc[0][mt][r] + biasA;
      const float gb = acc[1][mt][r] + biasB;
      const float gd = acc[2][mt][r] + biasD;
      const float Aval = 1.0f / (1.0f + __expf(-ga));            // sigmoid
      const float sp   = (gd > 20.0f) ? gd : log1pf(__expf(gd)); // softplus
      const float dtv  = sp * BASE_DT * scale;
      Ag[idx] = Aval;
      Dg[idx] = dtv;
      Bg[idx] = gb * dtv;
    }
  }
}

// ---------------- segmented scan, phase 1: per-segment summaries -------
__global__ __launch_bounds__(256)
void pm_scan_p1(const float* __restrict__ Ag, const float* __restrict__ Bg,
                const float* __restrict__ Dg,
                float* __restrict__ segP, float* __restrict__ segV,
                float* __restrict__ segPD, float* __restrict__ segVD) {
  const int tid  = blockIdx.x * blockDim.x + threadIdx.x;   // NCH*NSEG
  const int d    = tid % DIM;
  const int rest = tid / DIM;
  const int b    = rest % BATCH;
  const int s    = rest / BATCH;
  size_t idx = ((size_t)b * SEQ + (size_t)s * LSEG) * DIM + d;

  float P = 1.0f, v = 0.0f, spd = 0.0f, svd = 0.0f;
  for (int t = 0; t < LSEG; ++t, idx += DIM) {
    const float a  = Ag[idx];
    const float bb = Bg[idx];
    const float dt = Dg[idx];
    P *= a;
    v  = fmaf(a, v, bb);
    spd = fmaf(P, dt, spd);
    svd = fmaf(v, dt, svd);
  }
  segP[tid] = P; segV[tid] = v; segPD[tid] = spd; segVD[tid] = svd;
}

// ---------------- phase 2: combine 16 segment carries per channel ------
__global__ void pm_scan_p2(const float* __restrict__ segP,
                           const float* __restrict__ segV,
                           const float* __restrict__ segPD,
                           const float* __restrict__ segVD,
                           float* __restrict__ carryV,
                           float* __restrict__ carryX) {
  const int tid = blockIdx.x * blockDim.x + threadIdx.x;    // NCH threads
  float cv = 0.0f, cx = 0.0f;
#pragma unroll
  for (int s = 0; s < NSEG; ++s) {
    const int sidx = s * NCH + tid;
    carryV[sidx] = cv;
    carryX[sidx] = cx;
    cx += fmaf(cv, segPD[sidx], segVD[sidx]);
    cv  = fmaf(segP[sidx], cv, segV[sidx]);
  }
}

// ---------------- phase 3: replay with carries, emit outputs -----------
__global__ __launch_bounds__(256)
void pm_scan_p3(const float* __restrict__ Ag, const float* __restrict__ Bg,
                const float* __restrict__ Dg,
                const float* __restrict__ carryV,
                const float* __restrict__ carryX,
                float* __restrict__ x_seq, float* __restrict__ v_seq) {
  const int tid  = blockIdx.x * blockDim.x + threadIdx.x;   // NCH*NSEG
  const int d    = tid % DIM;
  const int rest = tid / DIM;
  const int b    = rest % BATCH;
  const int s    = rest / BATCH;
  size_t idx = ((size_t)b * SEQ + (size_t)s * LSEG) * DIM + d;

  float v = carryV[tid];
  float x = carryX[tid];
  for (int t = 0; t < LSEG; ++t, idx += DIM) {
    const float a  = Ag[idx];
    const float bb = Bg[idx];
    const float dt = Dg[idx];
    v = fmaf(a, v, bb);
    x = fmaf(v, dt, x);
    v_seq[idx] = v;
    x_seq[idx] = x;
  }
}

// ---------------- host-side launch -------------------------------------
extern "C" void kernel_launch(void* const* d_in, const int* in_sizes, int n_in,
                              void* d_out, int out_size, void* d_ws, size_t ws_size,
                              hipStream_t stream) {
  (void)in_sizes; (void)n_in; (void)out_size; (void)ws_size;
  // setup_inputs order: x, v, force, norm_x_w, norm_x_b, WA, bA, WB, bB, Wdt, bdt
  const float* force = (const float*)d_in[2];
  const float* WA    = (const float*)d_in[5];
  const float* bA    = (const float*)d_in[6];
  const float* WB    = (const float*)d_in[7];
  const float* bB    = (const float*)d_in[8];
  const float* Wdt   = (const float*)d_in[9];
  const float* bdt   = (const float*)d_in[10];

  float* x_seq = (float*)d_out;                          // [B,L,D]
  float* v_seq = x_seq + (size_t)MROWS * DIM;            // [B,L,D]

  // workspace carve-up
  char* p = (char*)d_ws;
  __bf16* Fbf  = (__bf16*)p; p += (size_t)MROWS * DIM * sizeof(__bf16);
  __bf16* WAbf = (__bf16*)p; p += (size_t)DIM * DIM * sizeof(__bf16);
  __bf16* WBbf = (__bf16*)p; p += (size_t)DIM * DIM * sizeof(__bf16);
  __bf16* Wdbf = (__bf16*)p; p += (size_t)DIM * DIM * sizeof(__bf16);
  float* Ag = (float*)p; p += (size_t)MROWS * DIM * sizeof(float);
  float* Bg = (float*)p; p += (size_t)MROWS * DIM * sizeof(float);
  float* Dg = (float*)p; p += (size_t)MROWS * DIM * sizeof(float);
  float* segP  = (float*)p; p += (size_t)NCH * NSEG * sizeof(float);
  float* segV  = (float*)p; p += (size_t)NCH * NSEG * sizeof(float);
  float* segPD = (float*)p; p += (size_t)NCH * NSEG * sizeof(float);
  float* segVD = (float*)p; p += (size_t)NCH * NSEG * sizeof(float);
  float* carryV = (float*)p; p += (size_t)NCH * NSEG * sizeof(float);
  float* carryX = (float*)p; p += (size_t)NCH * NSEG * sizeof(float);

  // 1) bf16 conversions (float4 -> 4x bf16)
  pm_cvt_bf16<<<2048, 256, 0, stream>>>((const float4*)force, (v4bf*)Fbf,
                                        (MROWS * DIM) / 4);
  pm_cvt_bf16<<<256, 256, 0, stream>>>((const float4*)WA,  (v4bf*)WAbf,
                                       (DIM * DIM) / 4);
  pm_cvt_bf16<<<256, 256, 0, stream>>>((const float4*)WB,  (v4bf*)WBbf,
                                       (DIM * DIM) / 4);
  pm_cvt_bf16<<<256, 256, 0, stream>>>((const float4*)Wdt, (v4bf*)Wdbf,
                                       (DIM * DIM) / 4);

  // 2) fused triple GEMM + gate epilogue (LDS double-buffered, async copy)
  dim3 ggrid(DIM / 64, MROWS / 64);          // 16 x 256 blocks
  pm_gemm3_epilogue<<<ggrid, 256, 0, stream>>>(Fbf, WAbf, WBbf, Wdbf,
                                               bA, bB, bdt, Ag, Bg, Dg);

  // 3) segmented affine scan + cumsum
  pm_scan_p1<<<(NCH * NSEG) / 256, 256, 0, stream>>>(Ag, Bg, Dg,
                                                     segP, segV, segPD, segVD);
  pm_scan_p2<<<NCH / 256, 256, 0, stream>>>(segP, segV, segPD, segVD,
                                            carryV, carryX);
  pm_scan_p3<<<(NCH * NSEG) / 256, 256, 0, stream>>>(Ag, Bg, Dg, carryV, carryX,
                                                     x_seq, v_seq);
}
